// SelfCorrelationBlock_71751723647296
// MI455X (gfx1250) — compile-verified, matrix operand order checked by reading, not run on hardware
//
#include <hip/hip_runtime.h>
#include <hip/hip_bf16.h>

typedef __attribute__((ext_vector_type(16))) _Float16 v16h;
typedef __attribute__((ext_vector_type(8)))  float    v8f;

#define BATCH 2
#define C     256
#define RC    64
#define HW    9216   // 96*96
#define KTOP  16
#define CHUNK 512    // columns per barrier pair (2 tiles per wave)

static __device__ inline unsigned umin32(unsigned a, unsigned b) { return a < b ? a : b; }
static __device__ inline unsigned umax32(unsigned a, unsigned b) { return a > b ? a : b; }

// Pack (float value, 14-bit index) into an order-preserving u32 key:
// ordered-float map in high 18 bits, column index in low 14 bits.
static __device__ inline unsigned pack_key(float v, int idx) {
  unsigned b = __float_as_uint(v);
  unsigned m = (unsigned)((int)b >> 31) | 0x80000000u;   // neg: ~0, pos: sign bit
  b ^= m;
  return (b & 0xFFFFC000u) | (unsigned)idx;
}

// Load 16 f16 values for one WMMA fragment register group following the
// CDNA5 16-bit A layout: halves [k0..k0+7] then [k0+16..k0+23], each 16B
// contiguous (row stride = 64 halves = 128B keeps 16B alignment).
static __device__ inline v16h load_frag16(const _Float16* p) {
  union { uint4 u[2]; v16h h; } f;
  f.u[0] = *reinterpret_cast<const uint4*>(p);
  f.u[1] = *reinterpret_cast<const uint4*>(p + 16);
  return f.h;
}

// Branch-free sorted insert into ascending top-K key list (drops current min).
// Pure median-of-3 network: 2 independent v_min_u32/v_max_u32 per slot,
// no compares, no select chains, no control flow.
static __device__ inline void topk_insert(unsigned (&keys)[KTOP], unsigned k) {
  #pragma unroll
  for (int t = 0; t < KTOP - 1; ++t)
    keys[t] = umax32(keys[t], umin32(k, keys[t + 1]));
  keys[KTOP - 1] = umax32(keys[KTOP - 1], k);
}

// ---------------- Kernel 1: reduce (1x1 conv) + L2-normalize ----------------
__global__ __launch_bounds__(256) void reduce_norm_kernel(
    const float* __restrict__ x, const float* __restrict__ w_reduce,
    float* __restrict__ nf32, _Float16* __restrict__ nf16)
{
  __shared__ float wsh[32 * C];          // 32 KB: half of w_reduce at a time
  const int tid = threadIdx.x;
  const int p   = blockIdx.x * 256 + tid;        // pixel over b*HW
  const int b   = p / HW;
  const int n   = p % HW;
  const float* xb = x + (size_t)b * C * HW + n;

  float acc[RC];
  #pragma unroll
  for (int r = 0; r < RC; ++r) acc[r] = 0.f;

  #pragma unroll
  for (int half = 0; half < 2; ++half) {
    __syncthreads();
    for (int i = tid; i < 32 * C; i += 256)
      wsh[i] = w_reduce[half * 32 * C + i];
    __syncthreads();
    for (int c = 0; c < C; ++c) {
      float xv = xb[(size_t)c * HW];             // coalesced across lanes
      #pragma unroll 8
      for (int r = 0; r < 32; ++r)
        acc[half * 32 + r] = fmaf(wsh[r * C + c], xv, acc[half * 32 + r]);
    }
  }

  float ss = 0.f;
  #pragma unroll
  for (int r = 0; r < RC; ++r) ss = fmaf(acc[r], acc[r], ss);
  float nrm = fmaxf(sqrtf(ss), 1e-12f);
  float inv = 1.0f / nrm;

  float*    o32 = nf32 + (size_t)p * RC;
  _Float16* o16 = nf16 + (size_t)p * RC;
  #pragma unroll
  for (int r = 0; r < RC; ++r) {
    float v = acc[r] * inv;
    o32[r] = v;
    o16[r] = (_Float16)v;
  }
}

// ------- Kernel 2: fused sim (WMMA) + top-k + gather-mean + proj + add ------
__global__ __launch_bounds__(512) void sim_topk_corr_proj_kernel(
    const _Float16* __restrict__ nf16, const float* __restrict__ nf32,
    const float* __restrict__ x, const float* __restrict__ w_proj,
    float* __restrict__ out)
{
  __shared__ float sim[16][CHUNK];       // 32 KB: one 16x512 sim stripe
  __shared__ int   topk[16][KTOP];       // 1 KB
  __shared__ float corr[16][RC];         // 4 KB

  const int tid  = threadIdx.x;
  const int lane = tid & 31;
  const int wave = tid >> 5;             // 0..15; wave w owns sim row w
  const int b       = blockIdx.x / (HW / 16);
  const int rowBase = (blockIdx.x % (HW / 16)) * 16;
  const size_t bN   = (size_t)b * HW;

  const int halfSel = (lane >= 16) ? 8 : 0;  // per-half-wave K split
  const int rlane   = lane & 15;

  // A fragments: 16 rows of nf (f16), reused for all 576 column tiles
  const _Float16* aRow = nf16 + (bN + rowBase + rlane) * RC;
  v16h a0 = load_frag16(aRow + 0  + halfSel);   // K = 0..31
  v16h a1 = load_frag16(aRow + 32 + halfSel);   // K = 32..63

  unsigned keys[KTOP];                   // lane-local sorted ascending keys
  #pragma unroll
  for (int t = 0; t < KTOP; ++t) keys[t] = 0u;   // 0 == -infinity sentinel

  const int rOff = (lane >= 16) ? 8 : 0;
  const float* simRow = sim[wave];

  for (int chunk = 0; chunk < HW / CHUNK; ++chunk) {   // 18 chunks
    const int colBase0 = chunk * CHUNK + wave * 16;
    const _Float16* bRow0 = nf16 + (bN + colBase0 + rlane) * RC;
    const _Float16* bRow1 = bRow0 + 256 * RC;          // second tile, +256 cols

    // prefetch next chunk's B rows (data is L2-resident; warms nearer levels)
    __builtin_prefetch(bRow0 + (size_t)CHUNK * RC, 0, 3);
    __builtin_prefetch(bRow1 + (size_t)CHUNK * RC, 0, 3);

    // issue all 8 b128 loads, then 4 WMMAs (loads overlap first WMMA)
    v16h b00 = load_frag16(bRow0 + 0  + halfSel);
    v16h b01 = load_frag16(bRow0 + 32 + halfSel);
    v16h b10 = load_frag16(bRow1 + 0  + halfSel);
    v16h b11 = load_frag16(bRow1 + 32 + halfSel);

    v8f acc0 = {};
    v8f acc1 = {};
    acc0 = __builtin_amdgcn_wmma_f32_16x16x32_f16(false, a0, false, b00,
                                                  (short)0, acc0, false, false);
    acc1 = __builtin_amdgcn_wmma_f32_16x16x32_f16(false, a0, false, b10,
                                                  (short)0, acc1, false, false);
    acc0 = __builtin_amdgcn_wmma_f32_16x16x32_f16(false, a1, false, b01,
                                                  (short)0, acc0, false, false);
    acc1 = __builtin_amdgcn_wmma_f32_16x16x32_f16(false, a1, false, b11,
                                                  (short)0, acc1, false, false);

    // C layout: VGPR v -> row v (lanes 0-15) / row v+8 (lanes 16-31), col = lane%16
    #pragma unroll
    for (int v = 0; v < 8; ++v) {
      sim[v + rOff][wave * 16 + rlane]       = acc0[v];
      sim[v + rOff][256 + wave * 16 + rlane] = acc1[v];
    }
    __syncthreads();

    // top-k scan: wave w scans row w; 16 candidates/lane as 8 float2 loads
    for (int j = 0; j < 8; ++j) {
      float2 sv = *reinterpret_cast<const float2*>(simRow + 2 * lane + 64 * j);
      int ci = chunk * CHUNK + 2 * lane + 64 * j;
      unsigned k0 = pack_key(sv.x, ci);
      unsigned k1 = pack_key(sv.y, ci + 1);
      if (umax32(k0, k1) > keys[0]) {    // single EXEC-guard; insert branch-free
        topk_insert(keys, k0);
        topk_insert(keys, k1);
      }
    }
    __syncthreads();
  }

  // Merge 32 lane-local top-16 key lists -> global top-16 for this row.
  // Keys are unique (low bits = column index), so the winner lane is the one
  // whose current max equals the wave max.
  for (int r = 0; r < KTOP; ++r) {
    unsigned mv = keys[KTOP - 1];
    #pragma unroll
    for (int s = 1; s < 32; s <<= 1)
      mv = umax32(mv, (unsigned)__shfl_xor((int)mv, s, 32));
    if (lane == 0) topk[wave][r] = (int)(mv & 0x3FFFu);
    if (keys[KTOP - 1] == mv) {          // unique winner pops its max
      #pragma unroll
      for (int t = KTOP - 1; t > 0; --t) keys[t] = keys[t - 1];
      keys[0] = 0u;
    }
  }
  __syncthreads();

  // corr = mean of top-16 nf rows (f32); lane handles components lane, lane+32
  {
    float c0 = 0.f, c1 = 0.f;
    #pragma unroll 4
    for (int t = 0; t < KTOP; ++t) {
      const float* p = nf32 + (bN + topk[wave][t]) * RC;
      c0 += p[lane];
      c1 += p[lane + 32];
    }
    corr[wave][lane]      = c0 * (1.0f / KTOP);
    corr[wave][lane + 32] = c1 * (1.0f / KTOP);
  }
  __syncthreads();

  // Projection to C=256 + residual: 16 rows * 256 ch = 4096 outputs, 8/thread
  #pragma unroll
  for (int j = 0; j < 8; ++j) {
    int o   = tid + 512 * j;
    int row = o & 15;
    int c   = o >> 4;
    const float* wp = w_proj + c * RC;
    const float* cr = corr[row];
    float e = 0.f;
    #pragma unroll 8
    for (int r2 = 0; r2 < RC; ++r2) e = fmaf(wp[r2], cr[r2], e);
    size_t oidx = ((size_t)b * C + c) * HW + rowBase + row;
    out[oidx] = x[oidx] + e;
  }
}

extern "C" void kernel_launch(void* const* d_in, const int* in_sizes, int n_in,
                              void* d_out, int out_size, void* d_ws, size_t ws_size,
                              hipStream_t stream) {
  const float* x        = (const float*)d_in[0];   // [2,256,96,96]
  const float* w_reduce = (const float*)d_in[1];   // [64,256]
  const float* w_proj   = (const float*)d_in[2];   // [256,64]
  float* out = (float*)d_out;

  // Workspace: nf16 (2*9216*64*2 = 2,359,296 B) then nf32 (4,718,592 B)
  char* ws = (char*)d_ws;
  _Float16* nf16 = (_Float16*)ws;
  float*    nf32 = (float*)(ws + 2359296);

  reduce_norm_kernel<<<(BATCH * HW) / 256, 256, 0, stream>>>(x, w_reduce, nf32, nf16);
  sim_topk_corr_proj_kernel<<<BATCH * (HW / 16), 512, 0, stream>>>(nf16, nf32, x, w_proj, out);
}
